// ManualLSTM_77919296684537
// MI455X (gfx1250) — compile-verified, matrix operand order checked by reading, not run on hardware
//
#include <hip/hip_runtime.h>
#include <hip/hip_fp16.h>

// ---------------------------------------------------------------------------
// ManualLSTM on MI455X (gfx1250, wave32, WMMA f16 -> f32 accumulate)
//
// Layouts chosen for minimal inner-loop address math:
//   CH[(T+1)][B][KTOT] : row = [x_t | h_{t-1}]  (A operand, single pointer)
//   W4[H][4][KTOT]     : 4 gate rows interleaved per hidden unit, so all 8
//                        weight loads per K-step use one address register
//                        with 24-bit immediate offsets (g*3072 + {0,32}).
// Per step: fused 4-gate GEMM via v_wmma_f32_16x16x32_f16, K split across
// wave pairs (LDS combine), unroll-2 register-pipelined loads, then
// sigmoid/tanh + c/h update in registers.  FC batched over all steps at end.
// ---------------------------------------------------------------------------

typedef __attribute__((ext_vector_type(16))) _Float16 v16h;
typedef __attribute__((ext_vector_type(8)))  _Float16 v8h;
typedef __attribute__((ext_vector_type(8)))  float    v8f;

#define B_   64
#define T_   512
#define I_   512
#define H_   1024
#define O_   512
#define KTOT (I_ + H_)      // 1536
#define KHALF 768           // K split per wave pair
#define NIT   (KHALF / 32)  // 24 WMMA K-steps per wave

union V16 { v16h v; v8h h[2]; };

// Fragment loader; `base` (0 or 8 elements) is pre-folded into `p`.
// lane L: row = L%16, K halves {base..base+7} and {base+16..base+23}.
__device__ __forceinline__ v16h load_frag(const _Float16* __restrict__ p) {
    V16 f;
    f.h[0] = *reinterpret_cast<const v8h*>(p);
    f.h[1] = *reinterpret_cast<const v8h*>(p + 16);
    return f.v;
}

__device__ __forceinline__ float sigmoidf_(float x) {
    return 1.0f / (1.0f + __expf(-x));
}

// ---------------------------------------------------------------------------
// Prep kernels
// ---------------------------------------------------------------------------

// Scatter x (f32 [B,T,I]) into CH rows' x-halves (f16).
__global__ void pack_x(_Float16* __restrict__ ch, const float* __restrict__ x, size_t n) {
    size_t i = (size_t)blockIdx.x * blockDim.x + threadIdx.x;
    size_t stride = (size_t)gridDim.x * blockDim.x;
    for (; i < n; i += stride) {
        const int b = (int)(i / ((size_t)T_ * I_));
        const int t = (int)((i / I_) % T_);
        const int k = (int)(i % I_);
        ch[((size_t)t * B_ + b) * KTOT + k] = (_Float16)x[i];
    }
}

// Interleave the 4 gate weight matrices: W4[j][g][k] = Wg[j][k]  (f32 -> f16)
__global__ void pack_w4(_Float16* __restrict__ w4,
                        const float* __restrict__ wf, const float* __restrict__ wi,
                        const float* __restrict__ wc, const float* __restrict__ wo,
                        size_t n) {   // n = H_*4*KTOT
    size_t i = (size_t)blockIdx.x * blockDim.x + threadIdx.x;
    size_t stride = (size_t)gridDim.x * blockDim.x;
    for (; i < n; i += stride) {
        const int j = (int)(i / (4 * KTOT));
        const int g = (int)((i / KTOT) & 3);
        const int k = (int)(i % KTOT);
        const size_t s = (size_t)j * KTOT + k;
        float v = (g == 0) ? wf[s] : (g == 1) ? wi[s] : (g == 2) ? wc[s] : wo[s];
        w4[i] = (_Float16)v;
    }
}

__global__ void cvt_f32_f16(_Float16* __restrict__ dst, const float* __restrict__ src, size_t n) {
    size_t i = (size_t)blockIdx.x * blockDim.x + threadIdx.x;
    size_t stride = (size_t)gridDim.x * blockDim.x;
    for (; i < n; i += stride) dst[i] = (_Float16)src[i];
}

// Zero CH[0]'s h-half and the c state.
__global__ void init_state(_Float16* __restrict__ ch, float* __restrict__ c, size_t n) {
    size_t i = (size_t)blockIdx.x * blockDim.x + threadIdx.x;
    size_t stride = (size_t)gridDim.x * blockDim.x;
    for (; i < n; i += stride) {
        const int b = (int)(i / H_);
        const int j = (int)(i % H_);
        ch[(size_t)b * KTOT + I_ + j] = (_Float16)0.0f;
        c[i] = 0.0f;
    }
}

// ---------------------------------------------------------------------------
// One LSTM timestep. grid = H_/16 = 64 blocks, block = 256 (8 waves).
// wave w: m-tile = (w&3)*16 (batch rows), k-half = w>>2.
// Wave pair (w, w+4): same 16x16 tile, disjoint K halves; combined via LDS.
// ---------------------------------------------------------------------------
__global__ void __launch_bounds__(256)
lstm_step_kernel(_Float16* __restrict__ ch,           // [(T+1),B,KTOT]
                 const _Float16* __restrict__ w4,     // [H,4,KTOT]
                 const float* __restrict__ bfp, const float* __restrict__ bip,
                 const float* __restrict__ bcp, const float* __restrict__ bop,
                 float* __restrict__ c_state,         // [B,H] f32
                 int t)
{
    __shared__ float red[4][32][33];   // padded: conflict-free

    const int lane  = threadIdx.x & 31;
    const int wave  = threadIdx.x >> 5;
    const int mw    = wave & 3;           // batch tile index
    const int khalf = wave >> 2;          // 0 or 1: K half
    const int m0    = mw * 16;
    const int j0    = blockIdx.x * 16;    // hidden-unit tile base
    const int r     = lane & 15;
    const int base  = (lane & 16) ? 8 : 0;
    const int k0    = khalf * KHALF;

    // Single A pointer: CH row (t, m0+r), our K half, base folded in.
    const _Float16* ap = ch + ((size_t)t * B_ + (m0 + r)) * KTOT + k0 + base;
    // Single W pointer: hidden row (j0+r); gates at immediate offsets g*KTOT.
    const _Float16* wp = w4 + ((size_t)(j0 + r) * 4) * KTOT + k0 + base;

    v8f accf = {}, acci = {}, accc = {}, acco = {};

    // ---- software-pipelined K loop (24 steps, unroll 2 -> no rotate copies) ----
    v16h a_c  = load_frag(ap);
    v16h b0_c = load_frag(wp + 0 * KTOT);
    v16h b1_c = load_frag(wp + 1 * KTOT);
    v16h b2_c = load_frag(wp + 2 * KTOT);
    v16h b3_c = load_frag(wp + 3 * KTOT);
    ap += 32; wp += 32;

    #pragma unroll 2
    for (int it = 0; it < NIT - 1; ++it) {
        v16h a_n  = load_frag(ap);
        v16h b0_n = load_frag(wp + 0 * KTOT);
        v16h b1_n = load_frag(wp + 1 * KTOT);
        v16h b2_n = load_frag(wp + 2 * KTOT);
        v16h b3_n = load_frag(wp + 3 * KTOT);
        ap += 32; wp += 32;
        accf = __builtin_amdgcn_wmma_f32_16x16x32_f16(false, a_c, false, b0_c, (short)0, accf, false, false);
        acci = __builtin_amdgcn_wmma_f32_16x16x32_f16(false, a_c, false, b1_c, (short)0, acci, false, false);
        accc = __builtin_amdgcn_wmma_f32_16x16x32_f16(false, a_c, false, b2_c, (short)0, accc, false, false);
        acco = __builtin_amdgcn_wmma_f32_16x16x32_f16(false, a_c, false, b3_c, (short)0, acco, false, false);
        a_c = a_n; b0_c = b0_n; b1_c = b1_n; b2_c = b2_n; b3_c = b3_n;
    }
    accf = __builtin_amdgcn_wmma_f32_16x16x32_f16(false, a_c, false, b0_c, (short)0, accf, false, false);
    acci = __builtin_amdgcn_wmma_f32_16x16x32_f16(false, a_c, false, b1_c, (short)0, acci, false, false);
    accc = __builtin_amdgcn_wmma_f32_16x16x32_f16(false, a_c, false, b2_c, (short)0, accc, false, false);
    acco = __builtin_amdgcn_wmma_f32_16x16x32_f16(false, a_c, false, b3_c, (short)0, acco, false, false);

    // ---- combine K halves via LDS ----
    if (khalf == 1) {
        #pragma unroll
        for (int q = 0; q < 8; ++q) {
            red[mw][lane][0 * 8 + q] = accf[q];
            red[mw][lane][1 * 8 + q] = acci[q];
            red[mw][lane][2 * 8 + q] = accc[q];
            red[mw][lane][3 * 8 + q] = acco[q];
        }
    }
    __syncthreads();
    if (khalf == 1) return;

    #pragma unroll
    for (int q = 0; q < 8; ++q) {
        accf[q] += red[mw][lane][0 * 8 + q];
        acci[q] += red[mw][lane][1 * 8 + q];
        accc[q] += red[mw][lane][2 * 8 + q];
        acco[q] += red[mw][lane][3 * 8 + q];
    }

    // ---- epilogue: lane owns column j = j0+r, rows m0+base .. m0+base+7 ----
    const int   j   = j0 + r;
    const float vbf = bfp[j], vbi = bip[j], vbc = bcp[j], vbo = bop[j];
    const int   mb  = m0 + base;
    #pragma unroll
    for (int q = 0; q < 8; ++q) {
        const int    b    = mb + q;
        const size_t cidx = (size_t)b * H_ + j;
        float fg = sigmoidf_(accf[q] + vbf);
        float ig = sigmoidf_(acci[q] + vbi);
        float cg = tanhf    (accc[q] + vbc);
        float og = sigmoidf_(acco[q] + vbo);
        float cn = fg * c_state[cidx] + ig * cg;
        c_state[cidx] = cn;
        float hn = og * tanhf(cn);
        // h goes into the h-half of the NEXT step's combined row
        ch[((size_t)(t + 1) * B_ + b) * KTOT + I_ + j] = (_Float16)hn;
    }
}

// ---------------------------------------------------------------------------
// Batched FC over all timesteps: Y[t,b,:] = h_t @ W_fc^T + b_fc
// grid = (O_/128, T_*B_/16), block = 256 (8 waves); wave w -> n tile.
// ---------------------------------------------------------------------------
__global__ void __launch_bounds__(256)
fc_kernel(const _Float16* __restrict__ ch,       // [(T+1),B,KTOT]; h at +I_
          const _Float16* __restrict__ Wfc,      // [O,H] f16
          const float* __restrict__ bfc,         // [O] f32
          float* __restrict__ ys)                // [B,T,O] f32
{
    const int lane = threadIdx.x & 31;
    const int wave = threadIdx.x >> 5;
    const int r    = lane & 15;
    const int base = (lane & 16) ? 8 : 0;

    const int n0 = (blockIdx.x * 8 + wave) * 16;   // output-col tile
    const int m0 = blockIdx.y * 16;                // row tile over t*B+b
    const int t  = m0 / B_;
    const int b0 = m0 % B_;

    const _Float16* ap = ch + ((size_t)(t + 1) * B_ + (b0 + r)) * KTOT + I_ + base;
    const _Float16* wp = Wfc + (size_t)(n0 + r) * H_ + base;

    v8f acc = {};
    v16h a_c = load_frag(ap);
    v16h w_c = load_frag(wp);
    ap += 32; wp += 32;
    #pragma unroll 2
    for (int kk = 32; kk < H_; kk += 32) {
        v16h a_n = load_frag(ap);
        v16h w_n = load_frag(wp);
        ap += 32; wp += 32;
        acc = __builtin_amdgcn_wmma_f32_16x16x32_f16(false, a_c, false, w_c, (short)0, acc, false, false);
        a_c = a_n; w_c = w_n;
    }
    acc = __builtin_amdgcn_wmma_f32_16x16x32_f16(false, a_c, false, w_c, (short)0, acc, false, false);

    const int   n    = n0 + r;
    const float bias = bfc[n];
    const int   bb   = b0 + base;
    #pragma unroll
    for (int q = 0; q < 8; ++q) {
        const int b = bb + q;
        ys[((size_t)b * T_ + t) * O_ + n] = acc[q] + bias;
    }
}

// final (h, c) -> tail of d_out
__global__ void finalize_kernel(const _Float16* __restrict__ ch,
                                const float* __restrict__ c_state,
                                float* __restrict__ out_tail)   // [2, B, H]
{
    const size_t n = (size_t)B_ * H_;
    size_t i = (size_t)blockIdx.x * blockDim.x + threadIdx.x;
    size_t stride = (size_t)gridDim.x * blockDim.x;
    for (; i < n; i += stride) {
        const int b = (int)(i / H_);
        const int j = (int)(i % H_);
        out_tail[i]     = (float)ch[((size_t)T_ * B_ + b) * KTOT + I_ + j];
        out_tail[n + i] = c_state[i];
    }
}

// ---------------------------------------------------------------------------
// kernel_launch
// d_in: 0:x 1:W_f 2:b_f 3:W_i 4:b_i 5:W_c 6:b_c 7:W_o 8:b_o 9:W_fc 10:b_fc
// ws: CH[(T+1)*B*KTOT] f16 (100.9MB) | W4[H*4*KTOT] f16 (12.6MB) |
//     Wfc[O*H] f16 (1MB) | c[B*H] f32 (0.25MB)   -> ~115 MB total
// ---------------------------------------------------------------------------
extern "C" void kernel_launch(void* const* d_in, const int* in_sizes, int n_in,
                              void* d_out, int out_size, void* d_ws, size_t ws_size,
                              hipStream_t stream) {
    const float* x    = (const float*)d_in[0];
    const float* Wf   = (const float*)d_in[1];
    const float* bf   = (const float*)d_in[2];
    const float* Wi   = (const float*)d_in[3];
    const float* bi   = (const float*)d_in[4];
    const float* Wc   = (const float*)d_in[5];
    const float* bc   = (const float*)d_in[6];
    const float* Wo   = (const float*)d_in[7];
    const float* bo   = (const float*)d_in[8];
    const float* Wfc  = (const float*)d_in[9];
    const float* bfc  = (const float*)d_in[10];
    float* out = (float*)d_out;

    const size_t ch_n  = (size_t)(T_ + 1) * B_ * KTOT; // 50,429,952
    const size_t w4_n  = (size_t)H_ * 4 * KTOT;        //  6,291,456
    const size_t wfc_n = (size_t)O_ * H_;              //    524,288
    const size_t bh_n  = (size_t)B_ * H_;

    _Float16* chp   = (_Float16*)d_ws;
    _Float16* w4p   = chp + ch_n;
    _Float16* wfcp  = w4p + w4_n;
    float*    cst   = (float*)(wfcp + wfc_n);

    // One-time layout/precision packing (deterministic; redone every call)
    pack_x     <<<2048, 256, 0, stream>>>(chp, x, (size_t)B_ * T_ * I_);
    pack_w4    <<<1024, 256, 0, stream>>>(w4p, Wf, Wi, Wc, Wo, w4_n);
    cvt_f32_f16<<<512,  256, 0, stream>>>(wfcp, Wfc, wfc_n);
    init_state <<<256,  256, 0, stream>>>(chp, cst, bh_n);

    // Sequential recurrence: one fused gate-GEMM + state-update kernel per step
    for (int t = 0; t < T_; ++t) {
        lstm_step_kernel<<<dim3(H_ / 16), dim3(256), 0, stream>>>(
            chp, w4p, bf, bi, bc, bo, cst, t);
    }

    // Batched output projection for all timesteps
    fc_kernel<<<dim3(O_ / 128, (T_ * B_) / 16), dim3(256), 0, stream>>>(
        chp, wfcp, bfc, out);

    // (h_final, c_final) after ys
    finalize_kernel<<<256, 256, 0, stream>>>(chp, cst, out + (size_t)B_ * T_ * O_);
}
